// MultiHeadAttention_2534030705095
// MI455X (gfx1250) — matrix operand order repaired and checked by reading, not compiled, not run on hardware
//
#include <hip/hip_runtime.h>
#include <hip/hip_bf16.h>

typedef __attribute__((ext_vector_type(16))) __bf16          v16bf;
typedef __attribute__((ext_vector_type(8)))  __bf16          v8bf;
typedef __attribute__((ext_vector_type(8)))  float           v8f;
typedef __attribute__((ext_vector_type(4)))  float           v4f;
typedef __attribute__((ext_vector_type(8)))  unsigned short  v8us;

#define B_  4
#define S_  2048
#define D_  1024
#define H_  16
#define HD_ 64
#define M_  (B_ * S_)   // 8192

__device__ __forceinline__ unsigned short f2bf(float f) {
  unsigned u = __builtin_bit_cast(unsigned, f);
  u += 0x7FFFu + ((u >> 16) & 1u);           // round-to-nearest-even
  return (unsigned short)(u >> 16);
}

__device__ __forceinline__ v8bf pack8(v4f a, v4f b) {
  v8us p;
  p[0] = f2bf(a[0]); p[1] = f2bf(a[1]); p[2] = f2bf(a[2]); p[3] = f2bf(a[3]);
  p[4] = f2bf(b[0]); p[5] = f2bf(b[1]); p[6] = f2bf(b[2]); p[7] = f2bf(b[3]);
  return __builtin_bit_cast(v8bf, p);
}

__device__ __forceinline__ v8f wmma_bf16(v16bf a, v16bf b, v8f c) {
  return __builtin_amdgcn_wmma_f32_16x16x32_bf16(
      false, a, false, b, (short)0, c, false, false);
}

union Frag { v16bf v; v8bf h[2]; };

// ---------------------------------------------------------------------------
// Weight pre-convert: f32 -> bf16 (lets the GEMM stage W tiles with async
// global->LDS copies, no per-tile conversion in the hot loop).
// ---------------------------------------------------------------------------
__global__ __launch_bounds__(256) void cvt_bf16(const float* __restrict__ src,
                                                unsigned short* __restrict__ dst) {
  const size_t i = ((size_t)blockIdx.x * blockDim.x + threadIdx.x) * 8;
  v4f a = *(const v4f*)(src + i);
  v4f b = *(const v4f*)(src + i + 4);
  *(v8us*)(dst + i) = __builtin_bit_cast(v8us, pack8(a, b));
}

// ---------------------------------------------------------------------------
// GEMM: Out = relu(X[M,K] * W[N,K]^T + bias[N]),  W pre-converted to bf16.
// MODE 0: bf16 out, head-split [B,H,S,HD]        (Q,K projections)
// MODE 1: bf16 out, head-split transposed [B,H,HD,S]  (V projection)
// MODE 2: f32  out, flat [M,N]                   (final projection)
// Block: 256 thr = 8 waves; tile 256(M) x 64(N); K-step 32; wave = 32 M-rows.
// W tile staged global->LDS via global_load_async_to_lds_b128 (ASYNCcnt).
// ---------------------------------------------------------------------------
template <bool A_BF16, int MODE>
__global__ __launch_bounds__(256) void gemm_bias_relu(
    const void* __restrict__ Xv, const unsigned short* __restrict__ Wb,
    const float* __restrict__ bias, void* __restrict__ Out) {
  constexpr int K = D_, N = D_;
  __shared__ alignas(16) unsigned short lb[64 * 40];  // 64x32 bf16 tile, 80B pitch

  const int tid  = threadIdx.x;
  const int lane = tid & 31;
  const int w    = tid >> 5;
  const int l16  = lane & 15;
  const int half = lane >> 4;
  const int koff = half * 8;
  const int n0   = blockIdx.x * 64;
  const int m0   = blockIdx.y * 256;

  // staging addresses (constant over the K loop except k offset)
  const int snl = tid >> 2;            // 0..63 : W row within tile
  const int skc = (tid & 3) * 8;       // 0,8,16,24 : k offset (8 bf16 = 16B)
  const unsigned lds_base = (unsigned)(uintptr_t)lb + (unsigned)(snl * 40 + skc) * 2u;
  const unsigned short* wsrc = Wb + (size_t)(n0 + snl) * K + skc;

  v8f acc[2][4];
#pragma unroll
  for (int mt = 0; mt < 2; ++mt)
#pragma unroll
    for (int nt = 0; nt < 4; ++nt) acc[mt][nt] = (v8f){0,0,0,0,0,0,0,0};

  for (int k0 = 0; k0 < K; k0 += 32) {
    __syncthreads();  // previous iteration's readers done
    // async stage of bf16 W tile into LDS (16B per lane)
    asm volatile("global_load_async_to_lds_b128 %0, %1, off"
                 :: "v"(lds_base), "v"(wsrc + k0) : "memory");
    asm volatile("s_wait_asynccnt 0" ::: "memory");
    __syncthreads();  // all waves' tile data visible

    // A fragments (2 M-tiles): lane half holds K {koff.., 16+koff..}
    Frag a[2];
#pragma unroll
    for (int mt = 0; mt < 2; ++mt) {
      const int arow = m0 + w * 32 + mt * 16 + l16;
      if constexpr (A_BF16) {
        const unsigned short* s0 = (const unsigned short*)Xv + (size_t)arow * K + k0;
        a[mt].h[0] = *(const v8bf*)(s0 + koff);
        a[mt].h[1] = *(const v8bf*)(s0 + 16 + koff);
      } else {
        const float* s0 = (const float*)Xv + (size_t)arow * K + k0;
        a[mt].h[0] = pack8(*(const v4f*)(s0 + koff),      *(const v4f*)(s0 + koff + 4));
        a[mt].h[1] = pack8(*(const v4f*)(s0 + 16 + koff), *(const v4f*)(s0 + 16 + koff + 4));
      }
    }

#pragma unroll
    for (int nt = 0; nt < 4; ++nt) {
      const int base = (nt * 16 + l16) * 40 + koff;  // B: lane=N, half holds K {koff.., 16+koff..}
      Frag b;
      b.h[0] = *(const v8bf*)&lb[base];
      b.h[1] = *(const v8bf*)&lb[base + 16];
      acc[0][nt] = wmma_bf16(a[0].v, b.v, acc[0][nt]);
      acc[1][nt] = wmma_bf16(a[1].v, b.v, acc[1][nt]);
    }
  }

  // epilogue: bias + relu + scatter
#pragma unroll
  for (int mt = 0; mt < 2; ++mt) {
#pragma unroll
    for (int nt = 0; nt < 4; ++nt) {
      const int n = n0 + nt * 16 + l16;
      const float bv = bias[n];
#pragma unroll
      for (int r = 0; r < 8; ++r) {
        const int m = m0 + w * 32 + mt * 16 + r + half * 8;
        float val = acc[mt][nt][r] + bv;
        val = val > 0.f ? val : 0.f;
        if constexpr (MODE == 2) {
          ((float*)Out)[(size_t)m * N + n] = val;
        } else {
          const int bi = m >> 11, s = m & (S_ - 1);
          const int h = n >> 6, hd = n & (HD_ - 1);
          const size_t idx = (MODE == 0)
              ? (((size_t)(bi * H_ + h) * S_ + s) * HD_ + hd)
              : (((size_t)(bi * H_ + h) * HD_ + hd) * S_ + s);
          ((unsigned short*)Out)[idx] = f2bf(val);
        }
      }
    }
  }
}

// ---------------------------------------------------------------------------
// Flash attention: one block per (b,h, 128 query rows); 8 waves x 16 q-rows.
// Key block = 64 keys. Row-sum via WMMA against a ones matrix; only the
// row-max uses the 16-lane butterfly.
// qh,kh: bf16 [B,H,S,HD]; vhT: bf16 [B,H,HD,S]; y out: bf16 [B,S,D].
// ---------------------------------------------------------------------------
__global__ __launch_bounds__(256) void flash_attn(
    const unsigned short* __restrict__ qh, const unsigned short* __restrict__ kh,
    const unsigned short* __restrict__ vhT, unsigned short* __restrict__ y) {
  __shared__ alignas(16) unsigned short pbuf[8 * 16 * 72];  // per-wave 16x64 P, 144B pitch

  const int tid  = threadIdx.x;
  const int lane = tid & 31;
  const int w    = tid >> 5;
  const int l16  = lane & 15;
  const int half = lane >> 4;
  const int koff = half * 8;
  const int bh   = blockIdx.y;
  const int bi   = bh >> 4, h = bh & 15;
  const int q0   = blockIdx.x * 128 + w * 16;

  const unsigned short* qbase = qh  + (size_t)bh * S_ * HD_;
  const unsigned short* kbase = kh  + (size_t)bh * S_ * HD_;
  const unsigned short* vbase = vhT + (size_t)bh * HD_ * S_;
  unsigned short* pw = &pbuf[w * 16 * 72];

  // ones B-fragment for WMMA row-sum
  Frag ones;
  {
    const v8us o8 = {0x3F80, 0x3F80, 0x3F80, 0x3F80, 0x3F80, 0x3F80, 0x3F80, 0x3F80};
    ones.h[0] = __builtin_bit_cast(v8bf, o8);
    ones.h[1] = __builtin_bit_cast(v8bf, o8);
  }

  // Q fragments for the two K-dim (=HD) steps
  Frag qa[2];
  {
    const unsigned short* qp = qbase + (size_t)(q0 + l16) * HD_;
#pragma unroll
    for (int t = 0; t < 2; ++t) {
      qa[t].h[0] = *(const v8bf*)(qp + t * 32 + koff);
      qa[t].h[1] = *(const v8bf*)(qp + t * 32 + 16 + koff);
    }
  }

  float mrun[8], lrun[8];
  v8f o[4];
#pragma unroll
  for (int r = 0; r < 8; ++r) { mrun[r] = -1e30f; lrun[r] = 0.f; }
#pragma unroll
  for (int nt = 0; nt < 4; ++nt) o[nt] = (v8f){0,0,0,0,0,0,0,0};

  for (int k0 = 0; k0 < S_; k0 += 64) {
    // ---- scores: 4 key-tiles x 2 HD-steps ----
    v8f sa[4];
#pragma unroll
    for (int nt = 0; nt < 4; ++nt) {
      const unsigned short* kp = kbase + (size_t)(k0 + nt * 16 + l16) * HD_ + koff;
      Frag b0, b1;
      b0.h[0] = *(const v8bf*)(kp);      b0.h[1] = *(const v8bf*)(kp + 16);
      b1.h[0] = *(const v8bf*)(kp + 32); b1.h[1] = *(const v8bf*)(kp + 48);
      v8f z = (v8f){0,0,0,0,0,0,0,0};
      z = wmma_bf16(qa[0].v, b0.v, z);
      sa[nt] = wmma_bf16(qa[1].v, b1.v, z);
    }

    // ---- online softmax: scale, row-max (butterfly), exp, P -> LDS ----
    float exf[8];
#pragma unroll
    for (int r = 0; r < 8; ++r) {
      float s0 = sa[0][r] * 0.125f, s1 = sa[1][r] * 0.125f;
      float s2 = sa[2][r] * 0.125f, s3 = sa[3][r] * 0.125f;
      float bm = fmaxf(fmaxf(s0, s1), fmaxf(s2, s3));
#pragma unroll
      for (int off = 1; off < 16; off <<= 1) bm = fmaxf(bm, __shfl_xor(bm, off, 32));
      const float mn = fmaxf(mrun[r], bm);
      exf[r] = __expf(mrun[r] - mn);
      mrun[r] = mn;
      const int ml = r + half * 8;
      pw[ml * 72 + l16]      = f2bf(__expf(s0 - mn));
      pw[ml * 72 + 16 + l16] = f2bf(__expf(s1 - mn));
      pw[ml * 72 + 32 + l16] = f2bf(__expf(s2 - mn));
      pw[ml * 72 + 48 + l16] = f2bf(__expf(s3 - mn));
    }
    asm volatile("s_wait_dscnt 0" ::: "memory");  // wave-local LDS RAW fence

    // ---- reload P as A-fragments (2 key-halves of 32) ----
    Frag pa[2];
    pa[0].h[0] = *(const v8bf*)&pw[l16 * 72 + koff];
    pa[0].h[1] = *(const v8bf*)&pw[l16 * 72 + 16 + koff];
    pa[1].h[0] = *(const v8bf*)&pw[l16 * 72 + 32 + koff];
    pa[1].h[1] = *(const v8bf*)&pw[l16 * 72 + 48 + koff];

    // ---- row-sum via WMMA with ones; update l ----
    {
      v8f rs = (v8f){0,0,0,0,0,0,0,0};
      rs = wmma_bf16(pa[0].v, ones.v, rs);
      rs = wmma_bf16(pa[1].v, ones.v, rs);
#pragma unroll
      for (int r = 0; r < 8; ++r) lrun[r] = lrun[r] * exf[r] + rs[r];
    }

    // ---- O = O*exf + P.V : 4 d-tiles x 2 key-halves ----
#pragma unroll
    for (int nt = 0; nt < 4; ++nt) {
      const unsigned short* vp = vbase + (size_t)(nt * 16 + l16) * S_ + k0 + koff;
      Frag vb0, vb1;
      vb0.h[0] = *(const v8bf*)(vp);      vb0.h[1] = *(const v8bf*)(vp + 16);
      vb1.h[0] = *(const v8bf*)(vp + 32); vb1.h[1] = *(const v8bf*)(vp + 48);
      v8f c;
#pragma unroll
      for (int r = 0; r < 8; ++r) c[r] = o[nt][r] * exf[r];
      c = wmma_bf16(pa[0].v, vb0.v, c);
      o[nt] = wmma_bf16(pa[1].v, vb1.v, c);
    }
  }

  // ---- normalize and merge heads into y[B,S,D] (bf16) ----
#pragma unroll
  for (int nt = 0; nt < 4; ++nt) {
#pragma unroll
    for (int r = 0; r < 8; ++r) {
      const float val = o[nt][r] / lrun[r];
      const int srow = q0 + r + half * 8;
      const int col  = h * HD_ + nt * 16 + l16;
      y[((size_t)bi * S_ + srow) * D_ + col] = f2bf(val);
    }
  }
}

// ---------------------------------------------------------------------------
extern "C" void kernel_launch(void* const* d_in, const int* in_sizes, int n_in,
                              void* d_out, int out_size, void* d_ws, size_t ws_size,
                              hipStream_t stream) {
  (void)in_sizes; (void)n_in; (void)out_size; (void)ws_size;
  const float* q  = (const float*)d_in[0];
  const float* k  = (const float*)d_in[1];
  const float* v  = (const float*)d_in[2];
  const float* Wq = (const float*)d_in[3];
  const float* bq = (const float*)d_in[4];
  const float* Wk = (const float*)d_in[5];
  const float* bk = (const float*)d_in[6];
  const float* Wv = (const float*)d_in[7];
  const float* bv = (const float*)d_in[8];
  const float* Wo = (const float*)d_in[9];
  const float* bo = (const float*)d_in[10];

  unsigned short* ws  = (unsigned short*)d_ws;
  const size_t NE = (size_t)M_ * D_;   // 8388608
  const size_t WE = (size_t)D_ * D_;   // 1048576
  unsigned short* qh  = ws;
  unsigned short* kh  = ws + NE;
  unsigned short* vhT = ws + 2 * NE;
  unsigned short* y   = ws + 3 * NE;
  unsigned short* Wqb = ws + 4 * NE;
  unsigned short* Wkb = Wqb + WE;
  unsigned short* Wvb = Wkb + WE;
  unsigned short* Wob = Wvb + WE;

  dim3 bb(256);
  dim3 gc(WE / (256 * 8));             // 512 blocks per weight convert
  cvt_bf16<<<gc, bb, 0, stream>>>(Wq, Wqb);
  cvt_bf16<<<gc, bb, 0, stream>>>(Wk, Wkb);
  cvt_bf16<<<gc, bb, 0, stream>>>(Wv, Wvb);
  cvt_bf16<<<gc, bb, 0, stream>>>(Wo, Wob);

  dim3 gg(D_ / 64, M_ / 256);          // (16, 32)
  gemm_bias_relu<false, 0><<<gg, bb, 0, stream>>>(q, Wqb, bq, qh);
  gemm_bias_relu<false, 0><<<gg, bb, 0, stream>>>(k, Wkb, bk, kh);
  gemm_bias_relu<false, 1><<<gg, bb, 0, stream>>>(v, Wvb, bv, vhT);
  flash_attn<<<dim3(S_ / 128, B_ * H_), bb, 0, stream>>>(qh, kh, vhT, y);
  gemm_bias_relu<true, 2><<<gg, bb, 0, stream>>>(y, Wob, bo, d_out);
}
